// generator_40149354283199
// MI455X (gfx1250) — compile-verified, matrix operand order checked by reading, not run on hardware
//
#include <hip/hip_runtime.h>
#include <hip/hip_bf16.h>
#include <math.h>

typedef __attribute__((ext_vector_type(16))) _Float16 v16h;
typedef __attribute__((ext_vector_type(8)))  _Float16 v8h;
typedef __attribute__((ext_vector_type(4)))  _Float16 v4h;
typedef __attribute__((ext_vector_type(8)))  float    v8f;

#define NNODES 8192
#define NEDGES 500000
#define PA 72   // LDS row pitch in halves (64 + 8 pad; 144B, 16B aligned, conflict-free)

// ---------------------------------------------------------------- utilities
__global__ void zero_f32(float* __restrict__ p, long long n) {
  long long i = (long long)blockIdx.x * blockDim.x + threadIdx.x;
  long long stride = (long long)gridDim.x * blockDim.x;
  for (; i < n; i += stride) p[i] = 0.0f;
}

__global__ void deg_count_kernel(const int* __restrict__ edges,
                                 float* __restrict__ outdeg,
                                 float* __restrict__ indeg) {
  long long t = (long long)blockIdx.x * blockDim.x + threadIdx.x;
  if (t >= 6LL * NEDGES) return;
  int r = (int)(t / NEDGES);
  int e = (int)(t % NEDGES);
  const int* base = edges + (long long)r * 2 * NEDGES;
  atomicAdd(&outdeg[r * NNODES + base[e]], 1.0f);
  atomicAdd(&indeg[r * NNODES + base[NEDGES + e]], 1.0f);
}

__global__ void deg_finalize_kernel(float* __restrict__ p, int n) {
  int i = blockIdx.x * blockDim.x + threadIdx.x;
  if (i < n) p[i] = rsqrtf(fmaxf(p[i], 1.0f));
}

// ------------------------------------------------------------ WMMA GEMM
// C[M x N] = act( scale_rows(A_concat) @ B + bias ), f16 WMMA, f32 accumulate.
// Block tile 64x64, 8 waves as 2(M) x 4(N); each wave: 32x16 strip (2 acc,
// shared B-fragment). K staged 64/chunk -> 4 WMMAs per wave per barrier pair.
// A staged row-major; B staged transposed so every fragment = 2x ds_load_b128.
template <int ACT, bool BIAS, bool SCALE, bool SPLIT>
__global__ void __launch_bounds__(256)
wmma_gemm_kernel(const float* __restrict__ A, int lda, int K1,
                 const float* __restrict__ A2, int lda2,
                 const float* __restrict__ B,
                 const float* __restrict__ bias,
                 const float* __restrict__ rowscale,
                 float* __restrict__ C, int N, int K) {
  __shared__ _Float16 As[64 * PA];   // As[row][k]
  __shared__ _Float16 BTs[64 * PA];  // BTs[col][k]
  const int tid  = threadIdx.x;
  const int lane = tid & 31;
  const int r16  = lane & 15;
  const int hs   = lane >> 4;           // K-half selector of the f16 layout
  const int w    = tid >> 5;
  const int wm   = w >> 2;              // 0..1 -> rows wm*32 .. wm*32+31
  const int wn   = w & 3;               // 0..3 -> cols wn*16 .. wn*16+15
  const int rowBlk = blockIdx.y * 64;
  const int colBlk = blockIdx.x * 64;

  v8f acc0 = {}, acc1 = {};

  for (int kk = 0; kk < K; kk += 64) {
    // ---- stage A (64 rows x 64 k): float4 loads -> f16x4 LDS stores ----
#pragma unroll
    for (int t = 0; t < 4; ++t) {
      int idx = t * 256 + tid;          // 1024 tasks = 64 rows x 16 float4
      int i  = idx >> 4;
      int j4 = (idx & 15) << 2;
      int gr = rowBlk + i, gc = kk + j4;
      float4 v;
      if (!SPLIT || gc < K1)
        v = *reinterpret_cast<const float4*>(A + (long long)gr * lda + gc);
      else
        v = *reinterpret_cast<const float4*>(A2 + (long long)gr * lda2 + (gc - K1));
      if (SCALE) { float s = rowscale[gr]; v.x *= s; v.y *= s; v.z *= s; v.w *= s; }
      v4h h = {(_Float16)v.x, (_Float16)v.y, (_Float16)v.z, (_Float16)v.w};
      *reinterpret_cast<v4h*>(&As[i * PA + j4]) = h;
    }
    // ---- stage B transposed (BTs[n][k]): 4 coalesced b32 loads / task ----
#pragma unroll
    for (int t = 0; t < 4; ++t) {
      int idx = t * 256 + tid;          // 1024 tasks = 16 kgroups x 64 cols
      int n  = idx & 63;                // coalesced across lanes
      int k0 = (idx >> 6) << 2;
      const float* bp = B + (long long)(kk + k0) * N + colBlk + n;
      v4h h = {(_Float16)bp[0], (_Float16)bp[(long long)N],
               (_Float16)bp[2LL * N], (_Float16)bp[3LL * N]};
      *reinterpret_cast<v4h*>(&BTs[n * PA + k0]) = h;
    }
    __syncthreads();

#pragma unroll
    for (int sub = 0; sub < 2; ++sub) {
      const int kb = sub * 32;
      // lane's 16 f16 elements = two contiguous 8-half runs (ds_load_b128 x2)
      const v8h a0lo = *(const v8h*)&As[(wm * 32 + r16) * PA + kb + hs * 8];
      const v8h a0hi = *(const v8h*)&As[(wm * 32 + r16) * PA + kb + 16 + hs * 8];
      const v8h a1lo = *(const v8h*)&As[(wm * 32 + 16 + r16) * PA + kb + hs * 8];
      const v8h a1hi = *(const v8h*)&As[(wm * 32 + 16 + r16) * PA + kb + 16 + hs * 8];
      const v8h blo  = *(const v8h*)&BTs[(wn * 16 + r16) * PA + kb + hs * 8];
      const v8h bhi  = *(const v8h*)&BTs[(wn * 16 + r16) * PA + kb + 16 + hs * 8];
      v16h af0 = __builtin_shufflevector(a0lo, a0hi, 0,1,2,3,4,5,6,7,8,9,10,11,12,13,14,15);
      v16h af1 = __builtin_shufflevector(a1lo, a1hi, 0,1,2,3,4,5,6,7,8,9,10,11,12,13,14,15);
      v16h bf  = __builtin_shufflevector(blo,  bhi,  0,1,2,3,4,5,6,7,8,9,10,11,12,13,14,15);
      acc0 = __builtin_amdgcn_wmma_f32_16x16x32_f16(false, af0, false, bf,
                                                    (short)0, acc0, false, false);
      acc1 = __builtin_amdgcn_wmma_f32_16x16x32_f16(false, af1, false, bf,
                                                    (short)0, acc1, false, false);
    }
    __syncthreads();
  }

  // ---- epilogue: C/D layout: VGPR v -> row v + 8*hs, lane%16 -> col ----
  const int col = colBlk + wn * 16 + r16;
  const int row0 = rowBlk + wm * 32 + 8 * hs;         // v = 0 row
  float* cp0 = C + (long long)row0 * N + col;          // rows row0 + v
  float* cp1 = cp0 + 16LL * N;                         // rows row0 + 16 + v
#pragma unroll
  for (int v = 0; v < 8; ++v) {
    float x0 = acc0[v];
    float x1 = acc1[v];
    if (BIAS) { const float bv = bias[col]; x0 += bv; x1 += bv; }
    if (ACT == 1) { x0 = fmaxf(x0, 0.0f); x1 = fmaxf(x1, 0.0f); }
    if (ACT == 2) {
      x0 = __builtin_amdgcn_rcpf(1.0f + __expf(-x0));
      x1 = __builtin_amdgcn_rcpf(1.0f + __expf(-x1));
    }
    *cp0 = x0;
    *cp1 = x1;
    cp0 += N;                                          // incremental addressing
    cp1 += N;
  }
}

// ------------------------------------------------------------ edge scatter
// feature dim D = 1<<dshift; one thread = one float4 chunk of one edge
__global__ void scatter_add_kernel(const float* __restrict__ xs,
                                   float* __restrict__ agg,
                                   const int* __restrict__ src,
                                   const int* __restrict__ dst,
                                   int dshift) {
  const int cshift = dshift - 2;
  long long t = (long long)blockIdx.x * blockDim.x + threadIdx.x;
  if (t >= ((long long)NEDGES << cshift)) return;
  int e = (int)(t >> cshift);
  int c = (int)(t & ((1 << cshift) - 1)) << 2;
  int s = src[e], d = dst[e];
  const float4 v = *reinterpret_cast<const float4*>(xs + ((long long)s << dshift) + c);
  float* o = agg + ((long long)d << dshift) + c;
  atomicAdd(o + 0, v.x);
  atomicAdd(o + 1, v.y);
  atomicAdd(o + 2, v.z);
  atomicAdd(o + 3, v.w);
}

__global__ void accum_kernel(float* __restrict__ tgt,
                             const float* __restrict__ agg,
                             const float* __restrict__ innorm,
                             const float* __restrict__ bias,
                             int D, int dshift) {
  long long t = (long long)blockIdx.x * blockDim.x + threadIdx.x;
  if (t >= ((long long)NNODES << dshift)) return;
  int n = (int)(t >> dshift);
  int f = (int)(t & (D - 1));
  tgt[t] += agg[t] * innorm[n] + bias[f];
}

// --------------------------------------------------- L1 normalize + slice
__global__ void l1_slice_kernel(const float* __restrict__ h,
                                const int* __restrict__ leftIdx,
                                float* __restrict__ fake) {
  __shared__ float s[128];
  int row = blockIdx.x;
  const float* x = h + (long long)(row + leftIdx[0]) * 128;
  float v = x[threadIdx.x];
  s[threadIdx.x] = fabsf(v);
  __syncthreads();
  for (int st = 64; st > 0; st >>= 1) {
    if ((int)threadIdx.x < st) s[threadIdx.x] += s[threadIdx.x + st];
    __syncthreads();
  }
  fake[(long long)row * 128 + threadIdx.x] = v / fmaxf(s[0], 1e-12f);
}

// ---------------------------------------------------------------- host side
static inline void launch_zero(float* p, long long n, hipStream_t s) {
  long long blocks = (n + 255) / 256;
  if (blocks > 65535) blocks = 65535;
  zero_f32<<<(int)blocks, 256, 0, s>>>(p, n);
}

template <int ACT, bool BIAS, bool SCALE, bool SPLIT>
static inline void launch_gemm(const float* A, int lda, int K1,
                               const float* A2, int lda2,
                               const float* B, const float* bias,
                               const float* rowscale, float* C,
                               int M, int N, int K, hipStream_t s) {
  dim3 grid(N / 64, M / 64);
  wmma_gemm_kernel<ACT, BIAS, SCALE, SPLIT>
      <<<grid, 256, 0, s>>>(A, lda, K1, A2, lda2, B, bias, rowscale, C, N, K);
}

extern "C" void kernel_launch(void* const* d_in, const int* in_sizes, int n_in,
                              void* d_out, int out_size, void* d_ws, size_t ws_size,
                              hipStream_t stream) {
  (void)in_sizes; (void)n_in; (void)out_size; (void)ws_size;
  const float* hM  = (const float*)d_in[0];
  const float* hD  = (const float*)d_in[1];
  const float* hT  = (const float*)d_in[2];
  const float* W1  = (const float*)d_in[3];
  const float* b1  = (const float*)d_in[4];
  const float* W2  = (const float*)d_in[5];
  const float* b2  = (const float*)d_in[6];
  const float* W3  = (const float*)d_in[7];
  const float* b3  = (const float*)d_in[8];
  const float* Adj = (const float*)d_in[9];
  const float* f1_w = (const float*)d_in[10];
  const float* f1_b = (const float*)d_in[11];
  const float* f2_w = (const float*)d_in[12];
  const float* f2_b = (const float*)d_in[13];
  const float* f3_w = (const float*)d_in[14];
  const float* f3_b = (const float*)d_in[15];
  const float* f4_w = (const float*)d_in[16];
  const float* f4_b = (const float*)d_in[17];
  const int* edges     = (const int*)d_in[18];
  const int* leftIndex = (const int*)d_in[20];
  float* out = (float*)d_out;

  const int N = NNODES;
  float* ws   = (float*)d_ws;
  float* outn = ws;                       // 6*N -> rsqrt(max(outdeg,1))
  float* inn  = outn + 6 * N;             // 6*N -> rsqrt(max(indeg,1))
  float* hA   = inn + 6 * N;              // 3 * N*128
  float* hB   = hA + 3LL * N * 128;       // 3 * N*128
  float* xs   = hB + 3LL * N * 128;       // N*128
  float* agg  = xs + (long long)N * 128;  // N*128
  // MLP temporaries overlay hB/xs/agg (dead after layer 3)
  float* x1 = hB;                         // 4096*256
  float* x2 = x1 + 4096LL * 256;          // 4096*512
  float* x3 = x2 + 4096LL * 512;          // 4096*1024

  // ---- degree norms ----
  launch_zero(outn, 12LL * N, stream);
  {
    long long tot = 6LL * NEDGES;
    deg_count_kernel<<<(int)((tot + 255) / 256), 256, 0, stream>>>(edges, outn, inn);
    int n = 12 * N;
    deg_finalize_kernel<<<(n + 255) / 256, 256, 0, stream>>>(outn, n);
  }

  const int srcIdx[6] = {0, 1, 0, 2, 2, 1};  // M,D,M,T,T,D
  const int tgtIdx[6] = {1, 0, 2, 0, 1, 2};  // newD,newM,newT,newM,newD,newT

  auto run_layer = [&](const float* const* inBuf, int Din,
                       const float* W, const float* bvec, int Dout, int dshift,
                       float* const* outBuf) {
    for (int i = 0; i < 3; ++i) launch_zero(outBuf[i], (long long)N * Dout, stream);
    for (int r = 0; r < 6; ++r) {
      launch_gemm<0, false, true, false>(inBuf[srcIdx[r]], Din, Din, nullptr, 0,
                                         W + (long long)r * Din * Dout, nullptr,
                                         outn + r * N, xs, N, Dout, Din, stream);
      launch_zero(agg, (long long)N * Dout, stream);
      const int* src = edges + (long long)r * 2 * NEDGES;
      const int* dst = src + NEDGES;
      long long cnt = (long long)NEDGES << (dshift - 2);
      scatter_add_kernel<<<(int)((cnt + 255) / 256), 256, 0, stream>>>(
          xs, agg, src, dst, dshift);
      long long tot = (long long)N << dshift;
      accum_kernel<<<(int)((tot + 255) / 256), 256, 0, stream>>>(
          outBuf[tgtIdx[r]], agg, inn + r * N, bvec + (long long)r * Dout,
          Dout, dshift);
    }
  };

  const float* l1in[3]  = {hM, hD, hT};
  float*       l1out[3] = {hA, hA + (long long)N * 128, hA + 2LL * N * 128};
  run_layer(l1in, 256, W1, b1, 64, 6, l1out);

  const float* l2in[3]  = {l1out[0], l1out[1], l1out[2]};
  float*       l2out[3] = {hB, hB + (long long)N * 128, hB + 2LL * N * 128};
  run_layer(l2in, 64, W2, b2, 64, 6, l2out);

  const float* l3in[3]  = {l2out[0], l2out[1], l2out[2]};
  float*       l3out[3] = {hA, hA + (long long)N * 128, hA + 2LL * N * 128};
  run_layer(l3in, 64, W3, b3, 128, 7, l3out);

  // fake embedding -> d_out[0 : 4096*128]
  l1_slice_kernel<<<4096, 128, 0, stream>>>(l3out[0], leftIndex, out);

  // ---- MLP ---- (f1 fuses concat([Adj | fake]) via split A-load at K1=8192)
  const float* fake = out;
  launch_gemm<1, true, false, true >(Adj, 8192, 8192, fake, 128, f1_w, f1_b,
                                     nullptr, x1, 4096, 256, 8320, stream);
  launch_gemm<1, true, false, false>(x1, 256, 256, nullptr, 0, f2_w, f2_b,
                                     nullptr, x2, 4096, 512, 256, stream);
  launch_gemm<1, true, false, false>(x2, 512, 512, nullptr, 0, f3_w, f3_b,
                                     nullptr, x3, 4096, 1024, 512, stream);
  launch_gemm<2, true, false, false>(x3, 1024, 1024, nullptr, 0, f4_w, f4_b,
                                     nullptr, out + 4096LL * 128, 4096, 8192,
                                     1024, stream);
}
